// MMEncoder_58497454571814
// MI455X (gfx1250) — matrix operand order repaired and checked by reading, not compiled, not run on hardware
//
#include <hip/hip_runtime.h>
#include <hip/hip_bf16.h>
#include <math.h>

// ---------------------------------------------------------------------------
// MI455X (gfx1250) implementation. Wave32, v_wmma_f32_16x16x32_bf16.
// All fragment traffic is 128-bit: A fragments exploit the contiguous-K runs
// of the documented A layout; B weights are pre-packed into fragment-major
// order; C tiles are staged through LDS for b128 row-major stores; the xg
// intermediates use a packed C-fragment layout end-to-end.
// ---------------------------------------------------------------------------

typedef __attribute__((ext_vector_type(16))) __bf16 v16bf;
typedef __attribute__((ext_vector_type(8)))  __bf16 v8bf;
typedef __attribute__((ext_vector_type(8)))  float  v8f;

#if __has_builtin(__builtin_amdgcn_tanhf)
__device__ __forceinline__ float fast_tanh(float x) { return __builtin_amdgcn_tanhf(x); }
#else
__device__ __forceinline__ float fast_tanh(float x) { return tanhf(x); }
#endif
__device__ __forceinline__ float sigf(float x) { return 1.0f / (1.0f + __expf(-x)); }

// --- A fragment (16x32 bf16, MxK). Lane layout (CDNA5 ISA §7.12.2):
// lanes 0-15: M=lane, K-base 0 ; lanes 16-31: M=lane-16, K-base +8.
// elems 0..7 -> K = 8*hlf + e (contiguous) ; elems 8..15 -> K = 16+8*hlf+e-8.
// => two contiguous 16B runs per lane: ideal for b128 loads.
__device__ __forceinline__ v16bf frag_a(const __bf16* __restrict__ rowp, int k0, int hlf) {
  const __bf16* p = rowp + k0 + (hlf << 3);
  v8bf lo = *(const v8bf*)p;
  v8bf hi = *(const v8bf*)(p + 16);
  return __builtin_shufflevector(lo, hi, 0, 1, 2, 3, 4, 5, 6, 7,
                                 8, 9, 10, 11, 12, 13, 14, 15);
}
__device__ __forceinline__ v16bf frag_a(const float* __restrict__ rowp, int k0, int hlf) {
  const float* p = rowp + k0 + (hlf << 3);
  float4 q0 = *(const float4*)p;
  float4 q1 = *(const float4*)(p + 4);
  float4 q2 = *(const float4*)(p + 16);
  float4 q3 = *(const float4*)(p + 20);
  v16bf a;
  a[0] = (__bf16)q0.x;  a[1] = (__bf16)q0.y;  a[2] = (__bf16)q0.z;  a[3] = (__bf16)q0.w;
  a[4] = (__bf16)q1.x;  a[5] = (__bf16)q1.y;  a[6] = (__bf16)q1.z;  a[7] = (__bf16)q1.w;
  a[8] = (__bf16)q2.x;  a[9] = (__bf16)q2.y;  a[10] = (__bf16)q2.z; a[11] = (__bf16)q2.w;
  a[12] = (__bf16)q3.x; a[13] = (__bf16)q3.y; a[14] = (__bf16)q3.z; a[15] = (__bf16)q3.w;
  return a;
}

// --- Packed-B layout: tile (nt,kc) of a [K,N] weight stored fragment-major:
// idx = ((nt*(K/32)+kc)*32 + lane)*16 + e, with
// n = nt*16 + (lane&15), k = kc*32 + 16*(e>=8) + 8*(lane>=16) + (e&7).
// GEMM B load = one contiguous 32B load per lane.
__device__ __forceinline__ v16bf frag_b(const __bf16* __restrict__ Wp, int ktiles,
                                        int nt, int kc, int lane) {
  return *(const v16bf*)(Wp + ((((size_t)nt * ktiles + kc) << 5 | lane) << 4));
}

__global__ void k_pack_b(const float* __restrict__ W, __bf16* __restrict__ Wp,
                         int K, int N) {
  int idx = blockIdx.x * blockDim.x + threadIdx.x;
  if (idx >= K * N) return;
  const int e = idx & 15;
  const int lane = (idx >> 4) & 31;
  const int tile = idx >> 9;
  const int ktiles = K >> 5;
  const int nt = tile / ktiles, kc = tile % ktiles;
  const int n = (nt << 4) + (lane & 15);
  const int k = (kc << 5) + ((e >> 3) << 4) + ((lane >> 4) << 3) + (e & 7);
  Wp[idx] = (__bf16)W[(size_t)k * N + n];
}

// ---------------------------------------------------------------------------
// Generic WMMA GEMM: C[M,N] = A[M,K] @ Wp + bias. One 16x16 tile per wave,
// 8 waves/block. Outputs (any non-null): Cf (f32 row-major via LDS stage),
// Cbf (bf16 row-major via LDS stage), Cpk (bf16 packed C-fragment layout).
template <typename TA>
__global__ void __launch_bounds__(256)
k_gemm(const TA* __restrict__ A, const __bf16* __restrict__ Wp,
       const float* __restrict__ bias,
       float* __restrict__ Cf, __bf16* __restrict__ Cbf, __bf16* __restrict__ Cpk,
       int M, int K, int N) {
  __shared__ __align__(32) float cstage[8 * 256];   // per-wave 16x16 f32 tile
  const int lane = threadIdx.x & 31;
  const int wave = threadIdx.x >> 5;
  const int hlf = lane >> 4;
  const int ntiles = N >> 4;
  const int ktiles = K >> 5;
  const long tile = (long)blockIdx.x * (blockDim.x >> 5) + wave;
  const long mt = tile / ntiles, nt = tile % ntiles;
  if (mt >= (M >> 4)) return;                       // wave-uniform
  const size_t m0 = (size_t)mt << 4, n0 = (size_t)nt << 4;

  v8f acc;
  const float bv = bias ? bias[n0 + (lane & 15)] : 0.0f;
#pragma unroll
  for (int r = 0; r < 8; ++r) acc[r] = bv;

  const TA* rowp = A + (m0 + (size_t)(lane & 15)) * K;
  for (int kc = 0; kc < ktiles; ++kc) {
    if (kc + 1 < ktiles)                            // -> global_prefetch_b8
      __builtin_prefetch(rowp + ((kc + 1) << 5), 0, 1);
    v16bf a = frag_a(rowp, kc << 5, hlf);
    v16bf b = frag_b(Wp, ktiles, (int)nt, kc, lane);
    acc = __builtin_amdgcn_wmma_f32_16x16x32_bf16(false, a, false, b,
                                                  (short)0, acc, false, false);
  }

  if (Cpk) {                                        // packed: contiguous b128 store
    v8bf v;
#pragma unroll
    for (int r = 0; r < 8; ++r) v[r] = (__bf16)acc[r];
    *(v8bf*)(Cpk + ((((size_t)mt * ntiles + nt) << 5 | lane) << 3)) = v;
  }
  if (Cf) {                                         // stage via LDS -> b128 rows
    float* st = cstage + (wave << 8);
#pragma unroll
    for (int r = 0; r < 8; ++r) st[((r + (hlf << 3)) << 4) + (lane & 15)] = acc[r];
    const int row = lane >> 1, c8 = (lane & 1) << 3;
    float* src = st + (row << 4) + c8;
    float* dst = Cf + (m0 + row) * N + n0 + c8;
    *(float4*)dst = *(const float4*)src;
    *(float4*)(dst + 4) = *(const float4*)(src + 4);
  }
  if (Cbf) {
    __bf16* stb = (__bf16*)(cstage + (wave << 8));
#pragma unroll
    for (int r = 0; r < 8; ++r) stb[((r + (hlf << 3)) << 4) + (lane & 15)] = (__bf16)acc[r];
    const int row = lane >> 1, c8 = (lane & 1) << 3;
    *(v8bf*)(Cbf + (m0 + row) * N + n0 + c8) = *(const v8bf*)(stb + (row << 4) + c8);
  }
}

// ---------------------------------------------------------------------------
// Four LSTM scans (mod{0,1} x fwd/bwd) as 4 concurrent workgroups.
// xg arrives in packed C-fragment layout (b128 per-lane loads); Wh is packed
// fragment-major; h lives in LDS (ds_read_b128 A fragments); hcat global
// stores are b128 rows read back from LDS after the step barrier.
__global__ void __launch_bounds__(1024)
k_lstm4(const __bf16* __restrict__ xgf0, const __bf16* __restrict__ xgb0,
        const __bf16* __restrict__ fh0,  const __bf16* __restrict__ bh0,
        __bf16* __restrict__ hc0,
        const __bf16* __restrict__ xgf1, const __bf16* __restrict__ xgb1,
        const __bf16* __restrict__ fh1,  const __bf16* __restrict__ bh1,
        __bf16* __restrict__ hc1,
        int T, int Bn, int Ha, int Hb) {
  __shared__ __align__(32) __bf16 hbuf[64 * 256];   // h_{t-1}: [B=64, H<=256]

  const int id = blockIdx.x;                        // 0..3
  const int m = id >> 1, rv = id & 1;
  const __bf16* xg = m ? (rv ? xgb1 : xgf1) : (rv ? xgb0 : xgf0);
  const __bf16* Wh = m ? (rv ? bh1 : fh1) : (rv ? bh0 : fh0);
  __bf16* hcat = m ? hc1 : hc0;
  const int H = m ? Hb : Ha;
  const int colOff = rv ? H : 0;
  const int H4 = 4 * H, H2 = 2 * H;
  const int ktiles = H >> 5;                        // K of recurrent GEMM
  const int xgtiles = H4 >> 4;                      // N tiles of xg rows

  const int lane = threadIdx.x & 31;
  const int wave = threadIdx.x >> 5;
  const int nwave = blockDim.x >> 5;                // 32
  const int hlf = lane >> 4;
  const int pairsTotal = (Bn >> 4) * (H >> 4);      // 64 / 32

  for (int i = threadIdx.x; i < Bn * H; i += blockDim.x) hbuf[i] = (__bf16)0.0f;

  float creg[2][8];
#pragma unroll
  for (int p = 0; p < 2; ++p)
#pragma unroll
    for (int r = 0; r < 8; ++r) creg[p][r] = 0.0f;

  __syncthreads();

  for (int t = 0; t < T; ++t) {
    const int tm = rv ? (T - 1 - t) : t;
    float hnew[2][8];
    int npair = 0;
    for (int pair = wave, p = 0; pair < pairsTotal; pair += nwave, ++p) {
      const int mt = pair & 3;                      // batch tile (B=64 -> 4)
      const int hs = pair >> 2;                     // h-strip
      const int m0 = mt << 4;
      v8f acc[4];
      // init with xg[t] (packed layout: one b128 load per gate tile)
#pragma unroll
      for (int g = 0; g < 4; ++g) {
        const size_t mtile = (size_t)tm * 4 + mt;
        const size_t ntile = (size_t)g * (H >> 4) + hs;
        v8bf xv = *(const v8bf*)(xg + (((mtile * xgtiles + ntile) << 5 | lane) << 3));
#pragma unroll
        for (int r = 0; r < 8; ++r) acc[g][r] = (float)xv[r];
      }
      // h_{t-1} @ Wh : A via ds_read_b128, B via packed 32B loads
      for (int kc = 0; kc < ktiles; ++kc) {
        const __bf16* hp = hbuf + (m0 + (lane & 15)) * H + (kc << 5) + (hlf << 3);
        v8bf lo = *(const v8bf*)hp;
        v8bf hi = *(const v8bf*)(hp + 16);
        v16bf a = __builtin_shufflevector(lo, hi, 0, 1, 2, 3, 4, 5, 6, 7,
                                          8, 9, 10, 11, 12, 13, 14, 15);
#pragma unroll
        for (int g = 0; g < 4; ++g) {
          v16bf b = frag_b(Wh, ktiles, g * (H >> 4) + hs, kc, lane);
          acc[g] = __builtin_amdgcn_wmma_f32_16x16x32_bf16(false, a, false, b,
                                                           (short)0, acc[g], false, false);
        }
      }
      // gates: c = tanh(a)*sig(i) + c*sig(f) ; h = tanh(c)*sig(o)
#pragma unroll
      for (int r = 0; r < 8; ++r) {
        const float av = acc[0][r], iv = acc[1][r], fv = acc[2][r], ov = acc[3][r];
        const float cn = fast_tanh(av) * sigf(iv) + creg[p][r] * sigf(fv);
        creg[p][r] = cn;
        hnew[p][r] = fast_tanh(cn) * sigf(ov);
      }
      npair = p + 1;
    }
    __syncthreads();                                // h_{t-1} reads done
    for (int pair = wave, p = 0; p < npair; pair += nwave, ++p) {
      const int mt = pair & 3;
      const int hs = pair >> 2;
      const int hcol = (hs << 4) + (lane & 15);
#pragma unroll
      for (int r = 0; r < 8; ++r)
        hbuf[((mt << 4) + r + (hlf << 3)) * H + hcol] = (__bf16)hnew[p][r];
    }
    __syncthreads();                                // hbuf == h_t
    // vectorized global store of h_t rows (b128 each)
    for (int c = threadIdx.x, nch = (Bn * H) >> 3; c < nch; c += blockDim.x) {
      const int row = c / (H >> 3);
      const int col8 = (c % (H >> 3)) << 3;
      *(v8bf*)(hcat + ((size_t)tm * Bn + row) * H2 + colOff + col8) =
          *(const v8bf*)(hbuf + row * H + col8);
    }
    // next write to hbuf happens only after next step's first barrier
  }
}

// ---------------------------------------------------------------------------
// ws[b,a] = s @ atW_w + atW_b  (tiny: 64x128)
__global__ void k_ws(const float* __restrict__ s, const float* __restrict__ Ww,
                     const float* __restrict__ Wb, float* __restrict__ ws,
                     int Bn, int S, int A) {
  int idx = blockIdx.x * blockDim.x + threadIdx.x;
  if (idx >= Bn * A) return;
  int b = idx / A, a = idx % A;
  float acc = Wb[a];
  for (int k = 0; k < S; ++k) acc += s[b * S + k] * Ww[k * A + a];
  ws[idx] = acc;
}

// logits[t,b] = tanh(vh[t,b,:] + ws[b,:]) . atw_w + atw_b   (float4 loads)
__global__ void k_logits(const float* __restrict__ vh, const float* __restrict__ ws,
                         const float* __restrict__ atw, const float* __restrict__ atb,
                         float* __restrict__ lg, int TB, int Bn, int A) {
  int idx = blockIdx.x * blockDim.x + threadIdx.x;
  if (idx >= TB) return;
  int b = idx % Bn;
  const float4* vp = (const float4*)(vh + (size_t)idx * A);
  const float4* wp = (const float4*)(ws + (size_t)b * A);
  const float4* ap = (const float4*)atw;
  float acc = atb[0];
  for (int q = 0; q < (A >> 2); ++q) {
    float4 v = vp[q], w = wp[q], aa = ap[q];
    acc += fast_tanh(v.x + w.x) * aa.x + fast_tanh(v.y + w.y) * aa.y +
           fast_tanh(v.z + w.z) * aa.z + fast_tanh(v.w + w.w) * aa.w;
  }
  lg[idx] = acc;
}

// Stable softmax over T, then ctx[b,:] = (1/T) sum_t a[t]*h[t,b,:]  (v8bf loads)
__global__ void __launch_bounds__(256)
k_ctx(const float* __restrict__ lg, const __bf16* __restrict__ hcat,
      __bf16* __restrict__ ctxbf, int T, int Bn, int C2) {
  __shared__ float red[256];
  __shared__ float aw[512];                         // T = 512
  const int b = blockIdx.x, tid = threadIdx.x;
  float mx = -1e30f;
  for (int t = tid; t < T; t += blockDim.x) mx = fmaxf(mx, lg[t * Bn + b]);
  red[tid] = mx; __syncthreads();
  for (int s = 128; s > 0; s >>= 1) {
    if (tid < s) red[tid] = fmaxf(red[tid], red[tid + s]);
    __syncthreads();
  }
  const float lmax = red[0]; __syncthreads();
  float ssum = 0.f;
  for (int t = tid; t < T; t += blockDim.x) {
    float e = __expf(lg[t * Bn + b] - lmax);
    aw[t] = e; ssum += e;
  }
  red[tid] = ssum; __syncthreads();
  for (int s = 128; s > 0; s >>= 1) {
    if (tid < s) red[tid] += red[tid + s];
    __syncthreads();
  }
  const float inv = 1.0f / (red[0] * (float)T);
  __syncthreads();
  const size_t rowstride = (size_t)Bn * C2;
  for (int c = tid, nch = C2 >> 3; c < nch; c += blockDim.x) {
    float acc[8] = {0, 0, 0, 0, 0, 0, 0, 0};
    const __bf16* hp = hcat + (size_t)b * C2 + (c << 3);
    for (int t = 0; t < T; ++t) {
      v8bf hv = *(const v8bf*)(hp + (size_t)t * rowstride);
      const float w = aw[t];
#pragma unroll
      for (int e = 0; e < 8; ++e) acc[e] += w * (float)hv[e];
    }
    v8bf o;
#pragma unroll
    for (int e = 0; e < 8; ++e) o[e] = (__bf16)(acc[e] * inv);
    *(v8bf*)(ctxbf + (size_t)b * C2 + (c << 3)) = o;
  }
}

// out = tanh(ctx0 @ lgd0 + b0 + ctx1 @ lgd1 + b1), [64,256] via WMMA
__global__ void __launch_bounds__(256)
k_out(const __bf16* __restrict__ c0, const __bf16* __restrict__ W0,
      const float* __restrict__ b0,
      const __bf16* __restrict__ c1, const __bf16* __restrict__ W1,
      const float* __restrict__ b1,
      float* __restrict__ out, int Bn, int K0, int K1, int N) {
  __shared__ __align__(32) float cstage[8 * 256];
  const int lane = threadIdx.x & 31;
  const int wave = threadIdx.x >> 5;
  const int hlf = lane >> 4;
  const int ntiles = N >> 4;
  const int tile = blockIdx.x * (blockDim.x >> 5) + wave;
  const int mt = tile / ntiles, nt = tile % ntiles;
  if (mt >= (Bn >> 4)) return;
  const size_t m0 = (size_t)mt << 4, n0 = (size_t)nt << 4;
  v8f acc;
  const float bv = b0[n0 + (lane & 15)] + b1[n0 + (lane & 15)];
#pragma unroll
  for (int r = 0; r < 8; ++r) acc[r] = bv;
  const __bf16* r0 = c0 + (m0 + (size_t)(lane & 15)) * K0;
  for (int kc = 0; kc < (K0 >> 5); ++kc) {
    v16bf a = frag_a(r0, kc << 5, hlf);
    v16bf b = frag_b(W0, K0 >> 5, nt, kc, lane);
    acc = __builtin_amdgcn_wmma_f32_16x16x32_bf16(false, a, false, b, (short)0, acc, false, false);
  }
  const __bf16* r1 = c1 + (m0 + (size_t)(lane & 15)) * K1;
  for (int kc = 0; kc < (K1 >> 5); ++kc) {
    v16bf a = frag_a(r1, kc << 5, hlf);
    v16bf b = frag_b(W1, K1 >> 5, nt, kc, lane);
    acc = __builtin_amdgcn_wmma_f32_16x16x32_bf16(false, a, false, b, (short)0, acc, false, false);
  }
  float* st = cstage + (wave << 8);
#pragma unroll
  for (int r = 0; r < 8; ++r)
    st[((r + (hlf << 3)) << 4) + (lane & 15)] = fast_tanh(acc[r]);
  const int row = lane >> 1, c8 = (lane & 1) << 3;
  float* dst = out + (m0 + row) * N + n0 + c8;
  const float* src = st + (row << 4) + c8;
  *(float4*)dst = *(const float4*)src;
  *(float4*)(dst + 4) = *(const float4*)(src + 4);
}

// ---------------------------------------------------------------------------
extern "C" void kernel_launch(void* const* d_in, const int* in_sizes, int n_in,
                              void* d_out, int out_size, void* d_ws, size_t ws_size,
                              hipStream_t stream) {
  (void)in_sizes; (void)n_in; (void)out_size; (void)ws_size;
  const int T = 512, B = 64, TB = T * B;
  const int AT = 128, S = 128, OUTN = 256;
  const int D[2] = {2048, 128}, P[2] = {512, 128}, H[2] = {256, 128};

  const float* x0  = (const float*)d_in[0];
  const float* x1  = (const float*)d_in[1];
  const float* sIn = (const float*)d_in[2];
  const float *emb_w[2], *emb_b[2], *fx_w[2], *fx_b[2], *fh_w[2];
  const float *bx_w[2], *bx_b[2], *bh_w[2];
  const float *atV_w[2], *atV_b[2], *atW_w[2], *atW_b[2], *atw_w[2], *atw_b[2];
  const float *lgd_w[2], *lgd_b[2];
  for (int m = 0; m < 2; ++m) {
    int o = 3 + 16 * m;
    emb_w[m] = (const float*)d_in[o + 0];  emb_b[m] = (const float*)d_in[o + 1];
    fx_w[m]  = (const float*)d_in[o + 2];  fx_b[m]  = (const float*)d_in[o + 3];
    fh_w[m]  = (const float*)d_in[o + 4];
    bx_w[m]  = (const float*)d_in[o + 5];  bx_b[m]  = (const float*)d_in[o + 6];
    bh_w[m]  = (const float*)d_in[o + 7];
    atV_w[m] = (const float*)d_in[o + 8];  atV_b[m] = (const float*)d_in[o + 9];
    atW_w[m] = (const float*)d_in[o + 10]; atW_b[m] = (const float*)d_in[o + 11];
    atw_w[m] = (const float*)d_in[o + 12]; atw_b[m] = (const float*)d_in[o + 13];
    lgd_w[m] = (const float*)d_in[o + 14]; lgd_b[m] = (const float*)d_in[o + 15];
  }

  // ---- workspace carve (~330 MB) ----
  char* base = (char*)d_ws;
  size_t off = 0;
  auto carve = [&](size_t elems, size_t esz) -> void* {
    off = (off + 255) & ~(size_t)255;
    void* p = base + off; off += elems * esz; return p;
  };
  __bf16 *embW[2], *fxW[2], *bxW[2], *fhW[2], *bhW[2], *atVW[2], *lgdW[2];
  __bf16 *h0[2], *xgf[2], *xgb[2], *hcat[2], *ctxb[2];
  float *vh[2], *wsb[2], *lg[2];
  for (int m = 0; m < 2; ++m) {
    embW[m] = (__bf16*)carve((size_t)D[m] * P[m], 2);
    fxW[m]  = (__bf16*)carve((size_t)P[m] * 4 * H[m], 2);
    bxW[m]  = (__bf16*)carve((size_t)P[m] * 4 * H[m], 2);
    fhW[m]  = (__bf16*)carve((size_t)H[m] * 4 * H[m], 2);
    bhW[m]  = (__bf16*)carve((size_t)H[m] * 4 * H[m], 2);
    atVW[m] = (__bf16*)carve((size_t)2 * H[m] * AT, 2);
    lgdW[m] = (__bf16*)carve((size_t)2 * H[m] * OUTN, 2);
    h0[m]   = (__bf16*)carve((size_t)TB * P[m], 2);
    xgf[m]  = (__bf16*)carve((size_t)TB * 4 * H[m], 2);   // packed layout
    xgb[m]  = (__bf16*)carve((size_t)TB * 4 * H[m], 2);   // packed layout
    hcat[m] = (__bf16*)carve((size_t)TB * 2 * H[m], 2);
    vh[m]   = (float*)carve((size_t)TB * AT, 4);
    wsb[m]  = (float*)carve((size_t)B * AT, 4);
    lg[m]   = (float*)carve((size_t)TB, 4);
    ctxb[m] = (__bf16*)carve((size_t)B * 2 * H[m], 2);
  }

  // 1) pack all WMMA B-operand weights into fragment-major bf16 (L2-resident)
  auto pack = [&](const float* w, __bf16* wp, int K, int N) {
    int n = K * N;
    k_pack_b<<<(n + 255) / 256, 256, 0, stream>>>(w, wp, K, N);
  };
  for (int m = 0; m < 2; ++m) {
    pack(emb_w[m], embW[m], D[m], P[m]);
    pack(fx_w[m],  fxW[m],  P[m], 4 * H[m]);
    pack(bx_w[m],  bxW[m],  P[m], 4 * H[m]);
    pack(fh_w[m],  fhW[m],  H[m], 4 * H[m]);
    pack(bh_w[m],  bhW[m],  H[m], 4 * H[m]);
    pack(atV_w[m], atVW[m], 2 * H[m], AT);
    pack(lgd_w[m], lgdW[m], 2 * H[m], OUTN);
  }

  // 2) h0 = x @ emb_w + emb_b   (bf16 row-major out)
  const float* xs[2] = {x0, x1};
  for (int m = 0; m < 2; ++m) {
    long tiles = (long)(TB / 16) * (P[m] / 16);
    k_gemm<float><<<(unsigned)(tiles / 8), 256, 0, stream>>>(
        xs[m], embW[m], emb_b[m], nullptr, h0[m], nullptr, TB, D[m], P[m]);
  }

  // 3) xg_{f,b} = h0 @ {fx,bx}_w + bias   (packed C-fragment bf16 out)
  for (int m = 0; m < 2; ++m) {
    long tiles = (long)(TB / 16) * (4 * H[m] / 16);
    k_gemm<__bf16><<<(unsigned)(tiles / 8), 256, 0, stream>>>(
        h0[m], fxW[m], fx_b[m], nullptr, nullptr, xgf[m], TB, P[m], 4 * H[m]);
    k_gemm<__bf16><<<(unsigned)(tiles / 8), 256, 0, stream>>>(
        h0[m], bxW[m], bx_b[m], nullptr, nullptr, xgb[m], TB, P[m], 4 * H[m]);
  }

  // 4) four LSTM scans, one workgroup each, concurrent
  k_lstm4<<<4, 1024, 0, stream>>>(xgf[0], xgb[0], fhW[0], bhW[0], hcat[0],
                                  xgf[1], xgb[1], fhW[1], bhW[1], hcat[1],
                                  T, B, H[0], H[1]);

  // 5) attention path
  for (int m = 0; m < 2; ++m) {
    long tiles = (long)(TB / 16) * (AT / 16);
    k_gemm<__bf16><<<(unsigned)(tiles / 8), 256, 0, stream>>>(
        hcat[m], atVW[m], atV_b[m], vh[m], nullptr, nullptr, TB, 2 * H[m], AT);
    k_ws<<<(B * AT + 255) / 256, 256, 0, stream>>>(sIn, atW_w[m], atW_b[m],
                                                   wsb[m], B, S, AT);
    k_logits<<<(TB + 255) / 256, 256, 0, stream>>>(vh[m], wsb[m], atw_w[m],
                                                   atw_b[m], lg[m], TB, B, AT);
    k_ctx<<<B, 256, 0, stream>>>(lg[m], hcat[m], ctxb[m], T, B, 2 * H[m]);
  }

  // 6) fusion + tanh -> d_out [64,256] f32
  {
    int tiles = (B / 16) * (OUTN / 16);             // 64 tiles, 8 blocks
    k_out<<<tiles / 8, 256, 0, stream>>>(ctxb[0], lgdW[0], lgd_b[0],
                                         ctxb[1], lgdW[1], lgd_b[1],
                                         (float*)d_out, B, 2 * H[0], 2 * H[1], OUTN);
  }
}
